// HybridDecoder_17892833755478
// MI455X (gfx1250) — compile-verified, matrix operand order checked by reading, not run on hardware
//
#include <hip/hip_runtime.h>

// ---------------- problem constants (match reference) ----------------
#define B_    1024
#define H_    1024
#define IN_   512
#define OUT_  8192
#define T_    15
#define EPS_  1e-5f

typedef __attribute__((ext_vector_type(16))) _Float16 v16h;
typedef __attribute__((ext_vector_type(8)))  _Float16 v8h;
typedef __attribute__((ext_vector_type(8)))  float    v8f;
typedef __attribute__((ext_vector_type(4)))  int      v4i;

#if __has_builtin(__builtin_amdgcn_global_load_async_to_lds_b128)
#define HAS_ASYNC_LDS 1
#else
#define HAS_ASYNC_LDS 0
#endif

#if HAS_ASYNC_LDS
typedef __attribute__((address_space(1))) v4i* gv4i_p;   // global int4*
typedef __attribute__((address_space(3))) v4i* lv4i_p;   // LDS int4*

__device__ __forceinline__ void async_copy_32B(const _Float16* g, _Float16* l)
{
    // two B128 async transfers: global -> LDS directly (ASYNCcnt), no VGPR round-trip
    __builtin_amdgcn_global_load_async_to_lds_b128((gv4i_p)(g), (lv4i_p)(l), 0, 0);
    __builtin_amdgcn_global_load_async_to_lds_b128((gv4i_p)(g), (lv4i_p)(l), 16, 0);
}
__device__ __forceinline__ void wait_async0()
{
#if __has_builtin(__builtin_amdgcn_s_wait_asynccnt)
    __builtin_amdgcn_s_wait_asynccnt(0);
#else
    asm volatile("s_wait_asynccnt 0x0" ::: "memory");
#endif
}
#endif

// ---------------- WMMA GEMM: C[M,N] = act( A(f32)[M,K] * W(f16)[N,K]^T + bias (+ C) )
// Block tile 128(M) x 128(N) x 32(K). 256 threads = 8 wave32 waves in a 2x4 grid,
// each wave owns a 64x32 region = 4x2 v_wmma_f32_16x16x32_f16 tiles (8 wmma / K-step).
// Double-buffered LDS; A staged via VGPRs (needs f32->f16 cvt), B staged via
// async direct-to-LDS DMA when the toolchain exposes it.
#define BM 128
#define BN 128
#define BK 32
#define LDSS 40   // padded LDS stride in halves (80B rows -> 16B aligned chunks)

template <int ACCUM, int ACT>
__global__ __launch_bounds__(256)
void gemm_f16w(const float* __restrict__ A, int lda,
               const _Float16* __restrict__ W,   // [N,K] row-major f16
               const float* __restrict__ bias,   // len N; ignored when ACCUM=1
               float* __restrict__ C, int ldc,
               int M, int N, int K)
{
    __shared__ _Float16 Ah[2][BM * LDSS];
    __shared__ _Float16 Bh[2][BN * LDSS];

    const int tid  = threadIdx.x;
    const int lane = tid & 31;
    const int wave = tid >> 5;
    const int bm   = blockIdx.y * BM;
    const int bn   = blockIdx.x * BN;
    const int wm   = (wave >> 2) * 64;   // 0 or 64
    const int wn   = (wave & 3)  * 32;   // 0,32,64,96

    // staging coordinates (same for A and B: 128 rows x 32 cols, 16 elems/thread)
    const int sr = tid >> 1;             // 0..127
    const int sc = (tid & 1) * 16;       // 0,16

    const float*    aptr = A + (size_t)(bm + sr) * lda + sc;
    const _Float16* wptr = W + (size_t)(bn + sr) * K   + sc;

    // prologue: stage K-tile 0 into buffer 0
    {
#if HAS_ASYNC_LDS
        async_copy_32B(wptr, &Bh[0][sr * LDSS + sc]);
#endif
        _Float16* ad = &Ah[0][sr * LDSS + sc];
        #pragma unroll
        for (int i = 0; i < 16; ++i) ad[i] = (_Float16)aptr[i];
#if HAS_ASYNC_LDS
        wait_async0();
#else
        _Float16* bd = &Bh[0][sr * LDSS + sc];
        #pragma unroll
        for (int i = 0; i < 16; ++i) bd[i] = wptr[i];
#endif
    }
    __syncthreads();

    v8f acc[4][2] = {};
    const int l  = lane & 15;
    const int ks = (lane >> 4) << 3;     // 0 or 8

    int buf = 0;
    for (int k0 = 0; k0 < K; k0 += BK, buf ^= 1) {
        const bool has_next = (k0 + BK) < K;

        // issue loads for the next K-tile NOW so they fly under the wmma chain
        float areg[16];
#if !HAS_ASYNC_LDS
        _Float16 breg[16];
#endif
        if (has_next) {
            if (k0 + 2 * BK < K) {
                __builtin_prefetch(aptr + k0 + 2 * BK, 0, 1);
                __builtin_prefetch(wptr + k0 + 2 * BK, 0, 1);
            }
#if HAS_ASYNC_LDS
            async_copy_32B(wptr + k0 + BK, &Bh[buf ^ 1][sr * LDSS + sc]);
#endif
            const float* as = aptr + k0 + BK;
            #pragma unroll
            for (int i = 0; i < 16; ++i) areg[i] = as[i];
#if !HAS_ASYNC_LDS
            const _Float16* bs = wptr + k0 + BK;
            #pragma unroll
            for (int i = 0; i < 16; ++i) breg[i] = bs[i];
#endif
        }

        // fragment loads per ISA 16-bit A/B layout:
        // lane<16: row = base+lane, halves = K[0..7] then K[16..23]
        // lane>=16: row = base+lane-16, halves = K[8..15] then K[24..31]
        v16h afrag[4], bfrag[2];
        #pragma unroll
        for (int i = 0; i < 4; ++i) {
            const v8h* ap = (const v8h*)&Ah[buf][(wm + i * 16 + l) * LDSS + ks];
            v8h lo = ap[0], hi = ap[2];          // +2 v8h = +16 halves
            #pragma unroll
            for (int t = 0; t < 8; ++t) { afrag[i][t] = lo[t]; afrag[i][t + 8] = hi[t]; }
        }
        #pragma unroll
        for (int j = 0; j < 2; ++j) {
            const v8h* bp = (const v8h*)&Bh[buf][(wn + j * 16 + l) * LDSS + ks];
            v8h lo = bp[0], hi = bp[2];
            #pragma unroll
            for (int t = 0; t < 8; ++t) { bfrag[j][t] = lo[t]; bfrag[j][t + 8] = hi[t]; }
        }
        #pragma unroll
        for (int i = 0; i < 4; ++i)
            #pragma unroll
            for (int j = 0; j < 2; ++j)
                acc[i][j] = __builtin_amdgcn_wmma_f32_16x16x32_f16(
                    false, afrag[i], false, bfrag[j], (short)0, acc[i][j], false, false);

        // stash next A tile into the other buffer (write-side never aliases read-side)
        if (has_next) {
            _Float16* ad = &Ah[buf ^ 1][sr * LDSS + sc];
            #pragma unroll
            for (int i = 0; i < 16; ++i) ad[i] = (_Float16)areg[i];
#if HAS_ASYNC_LDS
            wait_async0();                 // B tile landed in LDS before we publish
#else
            _Float16* bd = &Bh[buf ^ 1][sr * LDSS + sc];
            #pragma unroll
            for (int i = 0; i < 16; ++i) bd[i] = breg[i];
#endif
        }
        __syncthreads();
    }

    // epilogue: C/D layout -> lane<16: col=lane, row=base+r ; lane>=16: row=base+r+8
    const int moff = (lane >> 4) * 8;
    #pragma unroll
    for (int j = 0; j < 2; ++j) {
        const int col = bn + wn + j * 16 + l;
        const float bv = ACCUM ? 0.0f : bias[col];   // dead in ACCUM instantiation
        #pragma unroll
        for (int i = 0; i < 4; ++i) {
            const int row0 = bm + wm + i * 16 + moff;
            float* cp = C + (size_t)row0 * ldc + col;
            #pragma unroll
            for (int r = 0; r < 8; ++r) {
                float v = acc[i][j][r];
                if (ACCUM) v += *cp;                 // accumulate path: no +0 add
                else       v += bv;
                if (ACT == 1) v = fmaxf(v, 0.0f);    // relu
                *cp = v;
                cp += ldc;
            }
        }
    }
}

// ---------------- weight repack to f16 [N,K]: dst[n*K+k] = src[off + n*ns + k*ks]
__global__ void pack_f16_k(const float* __restrict__ src, _Float16* __restrict__ dst,
                           long total, int K, long ns, long ks, long off)
{
    long i = (long)blockIdx.x * blockDim.x + threadIdx.x;
    if (i >= total) return;
    long n = i / K;
    long k = i - n * K;
    dst[i] = (_Float16)src[off + n * ns + k * ks];
}

// ---------------- pointwise: x = BN(ELU(x)) over (t,b,H) with channel = i % H
__global__ void elu_bn_k(float* __restrict__ x, const float* __restrict__ g,
                         const float* __restrict__ bt, const float* __restrict__ m,
                         const float* __restrict__ v, long total)
{
    long i = (long)blockIdx.x * blockDim.x + threadIdx.x;
    if (i >= total) return;
    int c = (int)(i % H_);
    float val = x[i];
    float e = val > 0.f ? val : (expf(val) - 1.f);
    float inv = g[c] * rsqrtf(v[c] + EPS_);
    x[i] = (e - m[c]) * inv + bt[c];
}

// ---------------- GRU pointwise cell
__global__ void gru_point_k(const float* __restrict__ gi, const float* __restrict__ gh,
                            float* __restrict__ h, float* __restrict__ hs_t, int bh)
{
    int i = blockIdx.x * blockDim.x + threadIdx.x;
    if (i >= bh) return;
    int b = i / H_, j = i - b * H_;
    const float* gib = gi + (size_t)b * 3 * H_;
    const float* ghb = gh + (size_t)b * 3 * H_;
    float ir = gib[j], iz = gib[H_ + j], in = gib[2 * H_ + j];
    float hr = ghb[j], hz = ghb[H_ + j], hn = ghb[2 * H_ + j];
    float r  = 1.f / (1.f + expf(-(ir + hr)));
    float zg = 1.f / (1.f + expf(-(iz + hz)));
    float n  = tanhf(in + r * hn);
    float hv = (1.f - zg) * n + zg * h[i];
    h[i] = hv;
    hs_t[i] = hv;
}

__global__ void fill_k(float* __restrict__ p, float v, long n)
{
    long i = (long)blockIdx.x * blockDim.x + threadIdx.x;
    if (i < n) p[i] = v;
}

__global__ void copy_k(float* __restrict__ d, const float* __restrict__ s, long n)
{
    long i = (long)blockIdx.x * blockDim.x + threadIdx.x;
    if (i < n) d[i] = s[i];
}

// ---------------- host orchestration ----------------
extern "C" void kernel_launch(void* const* d_in, const int* in_sizes, int n_in,
                              void* d_out, int out_size, void* d_ws, size_t ws_size,
                              hipStream_t stream)
{
    const float* z        = (const float*)d_in[0];
    const float* cw[3]    = { (const float*)d_in[2], (const float*)d_in[4], (const float*)d_in[6] };
    const float* cb[3]    = { (const float*)d_in[3], (const float*)d_in[5], (const float*)d_in[7] };
    const float* bn1[4]   = { (const float*)d_in[8],  (const float*)d_in[9],
                              (const float*)d_in[10], (const float*)d_in[11] };
    const float* bn2[4]   = { (const float*)d_in[12], (const float*)d_in[13],
                              (const float*)d_in[14], (const float*)d_in[15] };
    const float* w_ih     = (const float*)d_in[16];
    const float* w_hh     = (const float*)d_in[17];
    const float* b_ih     = (const float*)d_in[18];
    const float* b_hh     = (const float*)d_in[19];
    const float* h2x_w    = (const float*)d_in[20];
    const float* h2x_b    = (const float*)d_in[21];
    const float* d2o_w    = (const float*)d_in[22];
    const float* d2o_b    = (const float*)d_in[23];
    const float* out_w    = (const float*)d_in[24];
    const float* out_b    = (const float*)d_in[25];

    // carve workspace
    char* wsp = (char*)d_ws;
    auto take = [&](size_t bytes) -> void* {
        void* p = (void*)wsp;
        wsp += (bytes + 255) & ~(size_t)255;
        return p;
    };
    const size_t BH = (size_t)B_ * H_;
    _Float16* cwT   = (_Float16*)take(9 * BH * sizeof(_Float16));           // 3 layers x 3 taps, [H,H]
    _Float16* wih_a = (_Float16*)take((size_t)3 * H_ * IN_ * sizeof(_Float16));
    _Float16* wih_b = (_Float16*)take((size_t)3 * H_ * H_  * sizeof(_Float16));
    _Float16* whh   = (_Float16*)take((size_t)3 * H_ * H_  * sizeof(_Float16));
    _Float16* h2xp  = (_Float16*)take((size_t)IN_ * H_     * sizeof(_Float16));
    _Float16* d2op  = (_Float16*)take((size_t)OUT_ * H_    * sizeof(_Float16));
    _Float16* outp  = (_Float16*)take((size_t)OUT_ * H_    * sizeof(_Float16));
    float* dec1 = (float*)take(3  * BH * sizeof(float));   // (t,b,h)
    float* dec2 = (float*)take(7  * BH * sizeof(float));
    float* dec3 = (float*)take(15 * BH * sizeof(float));
    float* hs   = (float*)take(15 * BH * sizeof(float));
    float* gi   = (float*)take((size_t)B_ * 3 * H_ * sizeof(float));
    float* gh   = (float*)take((size_t)B_ * 3 * H_ * sizeof(float));
    float* inp  = (float*)take((size_t)B_ * IN_ * sizeof(float));
    float* h    = (float*)take(BH * sizeof(float));

    auto gemm = [&](const float* A, int lda, const _Float16* W, const float* bias,
                    float* C, int ldc, int M, int N, int K, int accum, int act) {
        dim3 grid(N / BN, M / BM);
        if (accum)        gemm_f16w<1, 0><<<grid, dim3(256), 0, stream>>>(A, lda, W, bias, C, ldc, M, N, K);
        else if (act)     gemm_f16w<0, 1><<<grid, dim3(256), 0, stream>>>(A, lda, W, bias, C, ldc, M, N, K);
        else              gemm_f16w<0, 0><<<grid, dim3(256), 0, stream>>>(A, lda, W, bias, C, ldc, M, N, K);
    };
    auto pack = [&](const float* src, _Float16* dst, int N, int K, long ns, long ks, long off) {
        long total = (long)N * K;
        pack_f16_k<<<(int)((total + 255) / 256), 256, 0, stream>>>(src, dst, total, K, ns, ks, off);
    };

    // ---- pack all weights to f16 [N,K]
    for (int L = 0; L < 3; ++L)
        for (int k = 0; k < 3; ++k)   // Wk[o,i] = cw[(i*H + o)*3 + k]
            pack(cw[L], cwT + ((size_t)L * 3 + k) * BH, H_, H_, 3, 3 * H_, k);
    pack(w_ih, wih_a, 3 * H_, IN_, IN_ + H_, 1, 0);
    pack(w_ih, wih_b, 3 * H_, H_,  IN_ + H_, 1, IN_);
    pack(w_hh, whh,   3 * H_, H_,  H_, 1, 0);
    pack(h2x_w, h2xp, IN_,  H_, H_, 1, 0);
    pack(d2o_w, d2op, OUT_, H_, H_, 1, 0);
    pack(out_w, outp, OUT_, H_, H_, 1, 0);

    // ---- transposed conv stack: y[t] = sum_{2*ti+k==t} x[ti] @ Wk^T + bias
    auto convt = [&](const float* xin, int Lin, const _Float16* wp, const float* bias, float* yout) {
        int Lout = 2 * Lin + 1;
        for (int t = 0; t < Lout; ++t) {
            int first = 1;
            for (int k = 0; k < 3; ++k) {
                int ti2 = t - k;
                if (ti2 < 0 || (ti2 & 1)) continue;
                int ti = ti2 >> 1;
                if (ti >= Lin) continue;
                gemm(xin + (size_t)ti * BH, H_, wp + (size_t)k * BH,
                     first ? bias : nullptr, yout + (size_t)t * BH, H_,
                     B_, H_, H_, first ? 0 : 1, 0);
                first = 0;
            }
        }
    };
    convt(z,    1, cwT + 0 * BH, cb[0], dec1);
    { long n = 3 * (long)BH;
      elu_bn_k<<<(int)((n + 255) / 256), 256, 0, stream>>>(dec1, bn1[0], bn1[1], bn1[2], bn1[3], n); }
    convt(dec1, 3, cwT + 3 * BH, cb[1], dec2);
    { long n = 7 * (long)BH;
      elu_bn_k<<<(int)((n + 255) / 256), 256, 0, stream>>>(dec2, bn2[0], bn2[1], bn2[2], bn2[3], n); }
    convt(dec2, 7, cwT + 6 * BH, cb[2], dec3);   // no activation on layer 3

    // ---- GRU over T=15 steps (serial)
    copy_k<<<(int)((BH + 255) / 256), 256, 0, stream>>>(h, z, (long)BH);
    { long n = (long)B_ * IN_;
      fill_k<<<(int)((n + 255) / 256), 256, 0, stream>>>(inp, (float)(OUT_ - 2), n); }

    for (int t = 0; t < T_; ++t) {
        // gi = [inp, dec_t] @ w_ih^T + b_ih   (split concat into two GEMMs)
        gemm(inp, IN_, wih_a, b_ih, gi, 3 * H_, B_, 3 * H_, IN_, 0, 0);
        gemm(dec3 + (size_t)t * BH, H_, wih_b, nullptr, gi, 3 * H_, B_, 3 * H_, H_, 1, 0);
        // gh = h @ w_hh^T + b_hh
        gemm(h, H_, whh, b_hh, gh, 3 * H_, B_, 3 * H_, H_, 0, 0);
        // pointwise GRU cell -> h, hs[t]
        gru_point_k<<<(int)((BH + 255) / 256), 256, 0, stream>>>(gi, gh, h, hs + (size_t)t * BH, (int)BH);
        // inp = relu(h @ h2x_w^T + h2x_b)
        gemm(h, H_, h2xp, h2x_b, inp, IN_, B_, IN_, H_, 0, 1);
    }

    // ---- output projections: d_out = [ output(15360x8192) | dec_o(15360x8192) ]
    float* out0 = (float*)d_out;
    float* out1 = out0 + (size_t)T_ * B_ * OUT_;
    gemm(hs,   H_, outp, out_b, out0, OUT_, T_ * B_, OUT_, H_, 0, 0);
    gemm(dec3, H_, d2op, d2o_b, out1, OUT_, T_ * B_, OUT_, H_, 0, 0);
}
